// SpatialMamba_16106127360408
// MI455X (gfx1250) — matrix operand order repaired and checked by reading, not compiled, not run on hardware
//
#include <hip/hip_runtime.h>
#include <cstdint>
#include <cstddef>

// ---------------------------------------------------------------------------
// Bidirectional Mamba ("SpatialMamba") for MI455X / gfx1250, wave32 + WMMA.
//
// B = NM*T = 2048 sequences, L = V = 22 tokens, m = b*22 + l, TOK = 45056.
// DI = 256, D_STATE = 16.  xs[b,l,c] = x_flat[(nm*128+c)*1408 + (m-nm*1408)]:
// tokens contiguous per (nm,c) plane; 1408 = 16*88 so WMMA M-tiles never
// straddle planes.
// ---------------------------------------------------------------------------

typedef _Float16 v16h __attribute__((ext_vector_type(16)));
typedef _Float16 v8h  __attribute__((ext_vector_type(8)));
typedef float    v8f  __attribute__((ext_vector_type(8)));

// ---- CDNA5 async global->LDS path (ASYNCcnt), guarded for portability ----
#if __has_builtin(__builtin_amdgcn_global_load_async_to_lds_b128) && \
    __has_builtin(__builtin_amdgcn_s_wait_asynccnt)
#define USE_ASYNC_LDS 1
typedef int v4i_ld __attribute__((vector_size(16)));
typedef __attribute__((address_space(1))) v4i_ld* gv4i_p;   // global b128 ptr
typedef __attribute__((address_space(3))) v4i_ld* lv4i_p;   // LDS b128 ptr
#define AS_GLOBAL(p) ((gv4i_p)(uintptr_t)(p))
#define AS_LDS(p)    ((lv4i_p)(uint32_t)(uintptr_t)(p))
#endif

static __device__ __forceinline__ v16h mkfrag(v8h lo, v8h hi) {
  return __builtin_shufflevector(lo, hi, 0,1,2,3,4,5,6,7,8,9,10,11,12,13,14,15);
}
static __device__ __forceinline__ v8f wmma16(v16h a, v16h b, v8f c) {
  // D = A(16x32 f16) * B(32x16 f16) + C(16x16 f32)
  return __builtin_amdgcn_wmma_f32_16x16x32_f16(false, a, false, b,
                                                (short)0, c, false, false);
}
// A fragment from row-major [rows][ldk] f16: lane holds row (lane&15),
// k = ks*32 + 8*(lane>>4) + {0..7} and +16.
static __device__ __forceinline__ v16h load_afrag(const _Float16* p, int row,
                                                  int ldk, int ks, int hh) {
  const _Float16* q = p + (size_t)row * ldk + ks * 32 + hh * 8;
  v8h lo = *(const v8h*)q;
  v8h hi = *(const v8h*)(q + 16);
  return mkfrag(lo, hi);
}
// B fragment from row-major weights [n][K] (k-contiguous): lane holds column
// n (lane&15), k = ks*32 + 8*(lane>>4) + {0..7} and +16.
static __device__ __forceinline__ v16h load_bfrag(const _Float16* w, int nrow,
                                                  int ldk, int ks, int hh) {
  const _Float16* q = w + (size_t)nrow * ldk + ks * 32 + hh * 8;
  v8h lo = *(const v8h*)q;
  v8h hi = *(const v8h*)(q + 16);
  return mkfrag(lo, hi);
}

static __device__ __forceinline__ float fast_rcp(float x) {
#if __has_builtin(__builtin_amdgcn_rcpf)
  return __builtin_amdgcn_rcpf(x);      // v_rcp_f32, no div expansion
#else
  return 1.0f / x;
#endif
}
static __device__ __forceinline__ float siluf(float x) {
  return x * fast_rcp(1.0f + __expf(-x));
}

// ------------------------------- constants ---------------------------------
#define TOK   45056
#define NSEQ  2048
#define LQ    22
#define DIQ   256

// ----------------------------- K0: weight prep -----------------------------
__global__ void k_cvt_pad(const float* __restrict__ src, _Float16* __restrict__ dst,
                          int rows, int cols, int src_rows) {
  int idx = blockIdx.x * 256 + threadIdx.x;
  if (idx >= rows * cols) return;
  int r = idx / cols;
  dst[idx] = (r < src_rows) ? (_Float16)src[idx] : (_Float16)0.0f;
}

// W[c,j] = sum_k fuse_w[c, half+k] * out_w[k, j]   (128x256 result, f16)
__global__ void k_fusew(const float* __restrict__ fuse_w,
                        const float* __restrict__ out_w,
                        _Float16* __restrict__ Wd, int half) {
  int c = blockIdx.x;          // 0..127
  int j = threadIdx.x;         // 0..255
  float acc = 0.0f;
  for (int k = 0; k < 128; ++k)
    acc = fmaf(fuse_w[c * 256 + half + k], out_w[k * 256 + j], acc);
  Wd[c * 256 + j] = (_Float16)acc;
}

__global__ void k_aneg(const float* __restrict__ Alog, float* __restrict__ An) {
  int idx = blockIdx.x * 256 + threadIdx.x;   // 4096 = 256*16
  if (idx < DIQ * 16) An[idx] = -__expf(Alog[idx]);
}

// ------------------- K1: input projection GEMM (both dirs) -----------------
// out[m, n] = sum_k x[m,k] * in_w[n,k];  N-space = [f_xi|f_z|b_xi|b_z] = 1024
// xi stored f16; z stored as silu(z) f16. Backward tensors stored L-reversed.
__global__ __launch_bounds__(256) void k_inproj(
    const float* __restrict__ x,
    const _Float16* __restrict__ w_f, const _Float16* __restrict__ w_b,
    _Float16* __restrict__ xi_f, _Float16* __restrict__ xi_b,
    _Float16* __restrict__ zs_f, _Float16* __restrict__ zs_b) {
  __shared__ _Float16 ax[16 * 128];           // A tile [token][k] f16
  const int tid = threadIdx.x;
  const int m0 = blockIdx.x * 16;
  const int nm = m0 / 1408;
  const int toff = m0 - nm * 1408;

  { // stage: 16 contiguous tokens per channel plane, f32 -> f16 transpose
    int c = tid >> 1, half = tid & 1;
    const float* colp = x + (size_t)(nm * 128 + c) * 1408 + toff + half * 8;
    float4 p0 = *(const float4*)(colp);
    float4 p1 = *(const float4*)(colp + 4);
    int r = half * 8;
    ax[(r + 0) * 128 + c] = (_Float16)p0.x;
    ax[(r + 1) * 128 + c] = (_Float16)p0.y;
    ax[(r + 2) * 128 + c] = (_Float16)p0.z;
    ax[(r + 3) * 128 + c] = (_Float16)p0.w;
    ax[(r + 4) * 128 + c] = (_Float16)p1.x;
    ax[(r + 5) * 128 + c] = (_Float16)p1.y;
    ax[(r + 6) * 128 + c] = (_Float16)p1.z;
    ax[(r + 7) * 128 + c] = (_Float16)p1.w;
  }
  __syncthreads();

  const int lane = tid & 31;
  // wave id is uniform per wave: pin it to an SGPR so segment/dir selection
  // becomes scalar control flow (s_cbranch) instead of EXEC masking.
  const int wave = __builtin_amdgcn_readfirstlane(tid >> 5);
  const int mm = lane & 15, hh = lane >> 4;
  __builtin_prefetch(w_f + (size_t)wave * 8192, 0, 0);   // global_prefetch_b8

  v16h afr[4];
#pragma unroll
  for (int ks = 0; ks < 4; ++ks) afr[ks] = load_afrag(ax, mm, 128, ks, hh);

  // precompute reversed row index (for backward-direction stores)
  int mrev[8];
#pragma unroll
  for (int r = 0; r < 8; ++r) {
    int m = m0 + hh * 8 + r;
    int b = m / 22, l = m - b * 22;
    mrev[r] = b * 22 + 21 - l;
  }

  for (int nt8 = 0; nt8 < 8; ++nt8) {
    const int ntile = wave * 8 + nt8;          // 0..63 (scalar)
    const int dir   = ntile >> 5;              // 0=f, 1=b       (scalar)
    const bool is_z = ((ntile >> 4) & 1) != 0; // xi vs z        (scalar)
    const int nloc  = (ntile & 31) * 16 + mm;  // 0..511 within direction
    const int di    = (ntile & 15) * 16 + mm;  // 0..255 channel
    const _Float16* w = (dir ? w_b : w_f);
    v8f acc = {0.f, 0.f, 0.f, 0.f, 0.f, 0.f, 0.f, 0.f};
#pragma unroll
    for (int ks = 0; ks < 4; ++ks)
      acc = wmma16(afr[ks], load_bfrag(w, nloc, 128, ks, hh), acc);

    _Float16* dst = dir ? (is_z ? zs_b : xi_b) : (is_z ? zs_f : xi_f);
    if (is_z) {                                // scalar branch
#pragma unroll
      for (int r = 0; r < 8; ++r) {
        int mo = dir ? mrev[r] : (m0 + hh * 8 + r);
        dst[(size_t)mo * 256 + di] = (_Float16)siluf(acc[r]);
      }
    } else {
#pragma unroll
      for (int r = 0; r < 8; ++r) {
        int mo = dir ? mrev[r] : (m0 + hh * 8 + r);
        dst[(size_t)mo * 256 + di] = (_Float16)acc[r];
      }
    }
  }
}

// ---------- K2: causal conv + SiLU + xproj WMMA GEMM + dt + softplus -------
// One block = 64 tokens of one direction (launch per direction).
__global__ __launch_bounds__(256) void k_conv_proj(
    const _Float16* __restrict__ xi, const float* __restrict__ conv_w,
    const float* __restrict__ conv_b, const _Float16* __restrict__ w_xp,
    const float* __restrict__ dt_w, const float* __restrict__ dt_b,
    _Float16* __restrict__ xc, float* __restrict__ dbc,
    _Float16* __restrict__ dtq) {
  __shared__ _Float16 xcl[64 * 256];          // conv output tile, f16
  __shared__ float dbll[64 * 48];             // xproj output tile (40 padded)
  const int tid = threadIdx.x;
  const int m0 = blockIdx.x * 64;

  // phase 1: depthwise causal conv (K=4) + SiLU
  for (int idx = tid; idx < 64 * 256; idx += 256) {
    int ml = idx >> 8, di = idx & 255;
    int m = m0 + ml;
    int b = m / 22, l = m - b * 22;
    float a = conv_b[di];
#pragma unroll
    for (int k = 0; k < 4; ++k) {
      int lk = l - 3 + k;
      if (lk >= 0)
        a = fmaf((float)xi[(size_t)(m - 3 + k) * 256 + di], conv_w[di * 4 + k], a);
    }
    float s = siluf(a);
    xcl[idx] = (_Float16)s;
    xc[(size_t)m * 256 + di] = (_Float16)s;
  }
  __syncthreads();

  // phase 2: dbl = xc @ xproj_w.T  (M=64, N=48(pad of 40), K=256) via WMMA
  const int lane = tid & 31;
  const int wave = __builtin_amdgcn_readfirstlane(tid >> 5);   // scalar
  const int mm = lane & 15, hh = lane >> 4;
  for (int t = wave; t < 12; t += 8) {
    int mt = t / 3, nt = t % 3;
    v8f acc = {0.f, 0.f, 0.f, 0.f, 0.f, 0.f, 0.f, 0.f};
#pragma unroll
    for (int ks = 0; ks < 8; ++ks) {
      v16h a = load_afrag(xcl, mt * 16 + mm, 256, ks, hh);
      v16h b = load_bfrag(w_xp, nt * 16 + mm, 256, ks, hh);
      acc = wmma16(a, b, acc);
    }
#pragma unroll
    for (int r = 0; r < 8; ++r)
      dbll[(mt * 16 + hh * 8 + r) * 48 + nt * 16 + mm] = acc[r];
  }
  __syncthreads();

  // phase 3: store B(16) and C(16) rows (cols 8..39) as f32
  for (int idx = tid; idx < 64 * 32; idx += 256) {
    int ml = idx >> 5, j = idx & 31;
    dbc[(size_t)(m0 + ml) * 32 + j] = dbll[ml * 48 + 8 + j];
  }
  // phase 4: dt = softplus(dbl[:, :8] @ dt_w.T + dt_b)
  for (int idx = tid; idx < 64 * 256; idx += 256) {
    int ml = idx >> 8, di = idx & 255;
    float a = dt_b[di];
#pragma unroll
    for (int r = 0; r < 8; ++r)
      a = fmaf(dbll[ml * 48 + r], dt_w[di * 8 + r], a);
    float sp = (a > 20.0f) ? a : __logf(1.0f + __expf(a));  // v_log/v_exp
    dtq[(size_t)(m0 + ml) * 256 + di] = (_Float16)sp;
  }
}

// --------------------------- K3: selective scan ----------------------------
// One block per sequence; thread = channel di; h[16] in VGPRs; launch per dir.
__global__ __launch_bounds__(256) void k_scan(
    const float* __restrict__ An, const float* __restrict__ Dp,
    const float* __restrict__ dbc, const _Float16* __restrict__ dtq,
    const _Float16* __restrict__ xc, const _Float16* __restrict__ zs,
    _Float16* __restrict__ y) {
  __shared__ float bc[32];
  const int b = blockIdx.x;
  const int di = threadIdx.x;
  float a[16], h[16];
#pragma unroll
  for (int s = 0; s < 16; ++s) { a[s] = An[di * 16 + s]; h[s] = 0.0f; }
  const float dp = Dp[di];

  for (int l = 0; l < LQ; ++l) {
    if (di < 32) bc[di] = dbc[(size_t)(b * 22 + l) * 32 + di];
    __syncthreads();
    const size_t mi = (size_t)(b * 22 + l) * 256 + di;
    float dt = (float)dtq[mi];
    float u  = (float)xc[mi];
    float zz = (float)zs[mi];
    float dtu = dt * u;
    float yv = 0.0f;
#pragma unroll
    for (int s = 0; s < 16; ++s) {
      float e = __expf(dt * a[s]);             // v_exp_f32 (TRANS, co-issues)
      h[s] = fmaf(e, h[s], dtu * bc[s]);
      yv = fmaf(h[s], bc[16 + s], yv);
    }
    y[mi] = (_Float16)((yv + u * dp) * zz);
    __syncthreads();
  }
}

// ------------- K4: fused output GEMM (both dirs) + LayerNorm + store -------
__global__ __launch_bounds__(256) void k_out(
    const _Float16* __restrict__ yf, const _Float16* __restrict__ yb,
    const _Float16* __restrict__ Wf, const _Float16* __restrict__ Wb,
    const float* __restrict__ lnw, const float* __restrict__ lnb,
    float* __restrict__ out) {
  __shared__ _Float16 af[16 * 256];
  __shared__ _Float16 ab[16 * 256];
  __shared__ float ot[16 * 128];
  __shared__ float mu[16], rs[16];
  const int tid = threadIdx.x;
  const int m0 = blockIdx.x * 16;

  // stage forward rows (contiguous) and backward rows (un-reverse L)
#ifdef USE_ASYNC_LDS
  // CDNA5 async DMA: global -> LDS without VGPR round-trip (ASYNCcnt).
  for (int idx = tid; idx < 512; idx += 256)
    __builtin_amdgcn_global_load_async_to_lds_b128(
        AS_GLOBAL(yf + (size_t)m0 * 256 + idx * 8), AS_LDS(af + idx * 8), 0, 0);
  for (int idx = tid; idx < 512; idx += 256) {
    int r = idx >> 5, ch = idx & 31;
    int m = m0 + r;
    int b = m / 22, l = m - b * 22;
    __builtin_amdgcn_global_load_async_to_lds_b128(
        AS_GLOBAL(yb + (size_t)(b * 22 + 21 - l) * 256 + ch * 8),
        AS_LDS(ab + r * 256 + ch * 8), 0, 0);
  }
  __builtin_amdgcn_s_wait_asynccnt(0);         // s_wait_asynccnt 0
#else
  for (int idx = tid; idx < 512; idx += 256)
    ((uint4*)af)[idx] = ((const uint4*)(yf + (size_t)m0 * 256))[idx];
  for (int idx = tid; idx < 512; idx += 256) {
    int r = idx >> 5, ch = idx & 31;
    int m = m0 + r;
    int b = m / 22, l = m - b * 22;
    ((uint4*)(ab + r * 256))[ch] =
        *(const uint4*)(yb + (size_t)(b * 22 + 21 - l) * 256 + ch * 8);
  }
#endif
  __syncthreads();

  const int lane = tid & 31;
  const int wave = __builtin_amdgcn_readfirstlane(tid >> 5);   // scalar
  const int mm = lane & 15, hh = lane >> 4;
  const int nrow = wave * 16 + mm;            // output channel c, 0..127
  v8f acc = {0.f, 0.f, 0.f, 0.f, 0.f, 0.f, 0.f, 0.f};
#pragma unroll
  for (int ks = 0; ks < 8; ++ks)
    acc = wmma16(load_afrag(af, mm, 256, ks, hh),
                 load_bfrag(Wf, nrow, 256, ks, hh), acc);
#pragma unroll
  for (int ks = 0; ks < 8; ++ks)
    acc = wmma16(load_afrag(ab, mm, 256, ks, hh),
                 load_bfrag(Wb, nrow, 256, ks, hh), acc);
#pragma unroll
  for (int r = 0; r < 8; ++r)
    ot[(hh * 8 + r) * 128 + nrow] = acc[r];
  __syncthreads();

  if (tid < 16) {
    float s = 0.0f, s2 = 0.0f;
    for (int c = 0; c < 128; ++c) {
      float v = ot[tid * 128 + c];
      s += v; s2 = fmaf(v, v, s2);
    }
    float m_ = s * (1.0f / 128.0f);
    float var = s2 * (1.0f / 128.0f) - m_ * m_;
    mu[tid] = m_;
    rs[tid] = rsqrtf(var + 1e-5f);
  }
  __syncthreads();

  if (tid < 128) {
    int c = tid;
    int nm = m0 / 1408, toff = m0 - nm * 1408;
    float w = lnw[c], bb = lnb[c];
    float* op = out + (size_t)(nm * 128 + c) * 1408 + toff;
#pragma unroll
    for (int t = 0; t < 16; ++t)
      op[t] = (ot[t * 128 + c] - mu[t]) * rs[t] * w + bb;
  }
}

// ------------------------------ workspace map ------------------------------
#define SZ_WIN   (512 * 128 * 2)
#define SZ_WXP   (48 * 256 * 2)
#define SZ_WFU   (128 * 256 * 2)
#define SZ_AN    (256 * 16 * 4)
#define SZ_ACT16 ((size_t)TOK * 256 * 2)
#define SZ_DBC   ((size_t)TOK * 32 * 4)

#define OFF_WIN_F ((size_t)0)
#define OFF_WIN_B (OFF_WIN_F + SZ_WIN)
#define OFF_WXP_F (OFF_WIN_B + SZ_WIN)
#define OFF_WXP_B (OFF_WXP_F + SZ_WXP)
#define OFF_WF    (OFF_WXP_B + SZ_WXP)
#define OFF_WB    (OFF_WF + SZ_WFU)
#define OFF_AN_F  (OFF_WB + SZ_WFU)
#define OFF_AN_B  (OFF_AN_F + SZ_AN)
#define OFF_XI_F  (OFF_AN_B + SZ_AN)          // also reused as y_f (scan out)
#define OFF_XI_B  (OFF_XI_F + SZ_ACT16)       // also reused as y_b
#define OFF_ZS_F  (OFF_XI_B + SZ_ACT16)
#define OFF_ZS_B  (OFF_ZS_F + SZ_ACT16)
#define OFF_XC_F  (OFF_ZS_B + SZ_ACT16)
#define OFF_XC_B  (OFF_XC_F + SZ_ACT16)
#define OFF_DT_F  (OFF_XC_B + SZ_ACT16)
#define OFF_DT_B  (OFF_DT_F + SZ_ACT16)
#define OFF_DBC_F (OFF_DT_B + SZ_ACT16)
#define OFF_DBC_B (OFF_DBC_F + SZ_DBC)

extern "C" void kernel_launch(void* const* d_in, const int* in_sizes, int n_in,
                              void* d_out, int out_size, void* d_ws, size_t ws_size,
                              hipStream_t stream) {
  (void)in_sizes; (void)n_in; (void)out_size; (void)ws_size;
  const float* X        = (const float*)d_in[0];
  const float* f_in_w   = (const float*)d_in[1];
  const float* f_conv_w = (const float*)d_in[2];
  const float* f_conv_b = (const float*)d_in[3];
  const float* f_xproj  = (const float*)d_in[4];
  const float* f_dt_w   = (const float*)d_in[5];
  const float* f_dt_b   = (const float*)d_in[6];
  const float* f_Alog   = (const float*)d_in[7];
  const float* f_D      = (const float*)d_in[8];
  const float* f_out_w  = (const float*)d_in[9];
  const float* b_in_w   = (const float*)d_in[10];
  const float* b_conv_w = (const float*)d_in[11];
  const float* b_conv_b = (const float*)d_in[12];
  const float* b_xproj  = (const float*)d_in[13];
  const float* b_dt_w   = (const float*)d_in[14];
  const float* b_dt_b   = (const float*)d_in[15];
  const float* b_Alog   = (const float*)d_in[16];
  const float* b_D      = (const float*)d_in[17];
  const float* b_out_w  = (const float*)d_in[18];
  const float* fuse_w   = (const float*)d_in[19];
  const float* ln_w     = (const float*)d_in[20];
  const float* ln_b     = (const float*)d_in[21];

  char* ws = (char*)d_ws;
  _Float16* win_f = (_Float16*)(ws + OFF_WIN_F);
  _Float16* win_b = (_Float16*)(ws + OFF_WIN_B);
  _Float16* wxp_f = (_Float16*)(ws + OFF_WXP_F);
  _Float16* wxp_b = (_Float16*)(ws + OFF_WXP_B);
  _Float16* wfu_f = (_Float16*)(ws + OFF_WF);
  _Float16* wfu_b = (_Float16*)(ws + OFF_WB);
  float*    an_f  = (float*)(ws + OFF_AN_F);
  float*    an_b  = (float*)(ws + OFF_AN_B);
  _Float16* xi_f  = (_Float16*)(ws + OFF_XI_F);
  _Float16* xi_b  = (_Float16*)(ws + OFF_XI_B);
  _Float16* zs_f  = (_Float16*)(ws + OFF_ZS_F);
  _Float16* zs_b  = (_Float16*)(ws + OFF_ZS_B);
  _Float16* xc_f  = (_Float16*)(ws + OFF_XC_F);
  _Float16* xc_b  = (_Float16*)(ws + OFF_XC_B);
  _Float16* dt_f  = (_Float16*)(ws + OFF_DT_F);
  _Float16* dt_b  = (_Float16*)(ws + OFF_DT_B);
  float*    dbc_f = (float*)(ws + OFF_DBC_F);
  float*    dbc_b = (float*)(ws + OFF_DBC_B);
  _Float16* y_f = xi_f;   // xi dead after K2; scan output aliases it
  _Float16* y_b = xi_b;

  // K0: weight prep
  k_cvt_pad<<<(512 * 128 + 255) / 256, 256, 0, stream>>>(f_in_w, win_f, 512, 128, 512);
  k_cvt_pad<<<(512 * 128 + 255) / 256, 256, 0, stream>>>(b_in_w, win_b, 512, 128, 512);
  k_cvt_pad<<<(48 * 256 + 255) / 256, 256, 0, stream>>>(f_xproj, wxp_f, 48, 256, 40);
  k_cvt_pad<<<(48 * 256 + 255) / 256, 256, 0, stream>>>(b_xproj, wxp_b, 48, 256, 40);
  k_fusew<<<128, 256, 0, stream>>>(fuse_w, f_out_w, wfu_f, 0);
  k_fusew<<<128, 256, 0, stream>>>(fuse_w, b_out_w, wfu_b, 128);
  k_aneg<<<16, 256, 0, stream>>>(f_Alog, an_f);
  k_aneg<<<16, 256, 0, stream>>>(b_Alog, an_b);

  // K1: input projection (both directions fused in N dimension)
  k_inproj<<<TOK / 16, 256, 0, stream>>>(X, win_f, win_b, xi_f, xi_b, zs_f, zs_b);

  // K2: conv + SiLU + xproj GEMM + dt
  k_conv_proj<<<TOK / 64, 256, 0, stream>>>(xi_f, f_conv_w, f_conv_b, wxp_f,
                                            f_dt_w, f_dt_b, xc_f, dbc_f, dt_f);
  k_conv_proj<<<TOK / 64, 256, 0, stream>>>(xi_b, b_conv_w, b_conv_b, wxp_b,
                                            b_dt_w, b_dt_b, xc_b, dbc_b, dt_b);

  // K3: selective scan
  k_scan<<<NSEQ, 256, 0, stream>>>(an_f, f_D, dbc_f, dt_f, xc_f, zs_f, y_f);
  k_scan<<<NSEQ, 256, 0, stream>>>(an_b, b_D, dbc_b, dt_b, xc_b, zs_b, y_b);

  // K4: fused out+fuse GEMM, LayerNorm, transposed store
  k_out<<<TOK / 16, 256, 0, stream>>>(y_f, y_b, wfu_f, wfu_b, ln_w, ln_b,
                                      (float*)d_out);
}
